// ParallelRetention_origin_79508434583788
// MI455X (gfx1250) — matrix operand order repaired and checked by reading, not compiled, not run on hardware
//
#include <hip/hip_runtime.h>
#include <hip/hip_bf16.h>

typedef __attribute__((ext_vector_type(16))) __bf16 v16bf;
typedef __attribute__((ext_vector_type(8)))  float  v8f;

#define B_DIM 2
#define H_DIM 16
#define S_DIM 2048
#define D_DIM 64

union Frag { v16bf v; unsigned u32[8]; };

// alias-safe 32-bit LDS load from a bf16 array (uidx in 4-byte units)
__device__ __forceinline__ unsigned lds32(const __bf16* base, int uidx) {
  unsigned v;
  __builtin_memcpy(&v, (const char*)base + 4 * uidx, 4);
  return v;
}

// Pass 1: rs[h*S+q] = rsqrt(sum_k omask[h,q,k]), nan/inf -> 0
__global__ __launch_bounds__(256)
void ParallelRetention_rowscale(const float* __restrict__ omask, float* __restrict__ rs) {
  const int row  = blockIdx.x * 8 + (threadIdx.x >> 5);
  const int lane = threadIdx.x & 31;
  const float* p = omask + (size_t)row * S_DIM;
  float s = 0.f;
  for (int i = 0; i < 16; ++i) {
    float4 f = *(const float4*)(p + i * 128 + lane * 4);
    s += f.x + f.y + f.z + f.w;
  }
  for (int msk = 16; msk; msk >>= 1) s += __shfl_xor(s, msk);
  if (lane == 0) {
    float sc = rsqrtf(s);
    if (!__builtin_isfinite(sc)) sc = 0.f;
    rs[row] = sc;
  }
}

// Pass 2: streaming retention. One workgroup = (b,h) x 128 query rows.
__global__ __launch_bounds__(256)
void ParallelRetention_main(const float* __restrict__ qg, const float* __restrict__ kg,
                            const float* __restrict__ vg, const float* __restrict__ omask,
                            const float* __restrict__ rs, float* __restrict__ out) {
  __shared__ __bf16 lq[128 * 68];  // Q block (pre-scaled by rs), bf16, padded rows
  __shared__ __bf16 lk[32 * 68];   // K chunk, row-major
  __shared__ __bf16 lv[64 * 36];   // V chunk, transposed [d][k]
  __shared__ __bf16 lp[128 * 34];  // per-wave masked-score scratch

  const int tid  = threadIdx.x;
  const int lane = tid & 31;
  const int w    = tid >> 5;
  const int m    = lane & 15;
  const int half = lane >> 4;
  const int bh   = blockIdx.y;
  const int h    = bh % H_DIM;
  const int qb   = blockIdx.x * 128;

  const size_t bhBase = (size_t)bh * S_DIM * D_DIM;
  const float* qp = qg + bhBase;
  const float* kp = kg + bhBase;
  const float* vp = vg + bhBase;

  // stage Q [128 x 64] -> bf16 LDS, pre-scaled by per-row decay normalizer rs[q]
  for (int i = 0; i < 8; ++i) {
    int idx4 = i * 256 + tid;                // 2048 float4
    int row  = idx4 >> 4;
    int c4   = (idx4 & 15) * 4;
    float sc = rs[h * S_DIM + qb + row];
    float4 f = *(const float4*)(qp + (size_t)(qb + row) * D_DIM + c4);
    __bf16* d = &lq[row * 68 + c4];
    d[0] = (__bf16)(f.x * sc); d[1] = (__bf16)(f.y * sc);
    d[2] = (__bf16)(f.z * sc); d[3] = (__bf16)(f.w * sc);
  }
  __syncthreads();

  // per-lane Q A-fragments (documented 16-bit A layout), kept for whole k-loop
  Frag aQ0, aQ1;
  {
    const __bf16* row = &lq[(w * 16 + m) * 68];
    for (int j = 0; j < 4; ++j) {
      aQ0.u32[j]     = lds32(row, half * 4 + j);            // d 0..15 region
      aQ0.u32[4 + j] = lds32(row, 8 + half * 4 + j);        // d 16..31 region
      aQ1.u32[j]     = lds32(row, 16 + half * 4 + j);       // d 32..47
      aQ1.u32[4 + j] = lds32(row, 24 + half * 4 + j);       // d 48..63
    }
  }

  v8f acc[4];
  for (int dt = 0; dt < 4; ++dt)
    for (int i = 0; i < 8; ++i) acc[dt][i] = 0.f;
  float srow[8] = {0.f, 0.f, 0.f, 0.f, 0.f, 0.f, 0.f, 0.f};

  for (int kb = 0; kb < S_DIM; kb += 32) {
    __syncthreads();
    // stage K chunk (row-major) and V chunk (transposed) as bf16
    for (int i = 0; i < 2; ++i) {
      int idx4 = i * 256 + tid;              // 512 float4 (32 rows x 16)
      int row  = idx4 >> 4;
      int c4   = (idx4 & 15) * 4;
      float4 fk = *(const float4*)(kp + (size_t)(kb + row) * D_DIM + c4);
      __bf16* dk = &lk[row * 68 + c4];
      dk[0] = (__bf16)fk.x; dk[1] = (__bf16)fk.y; dk[2] = (__bf16)fk.z; dk[3] = (__bf16)fk.w;
      float4 fv = *(const float4*)(vp + (size_t)(kb + row) * D_DIM + c4);
      lv[(c4 + 0) * 36 + row] = (__bf16)fv.x;
      lv[(c4 + 1) * 36 + row] = (__bf16)fv.y;
      lv[(c4 + 2) * 36 + row] = (__bf16)fv.z;
      lv[(c4 + 3) * 36 + row] = (__bf16)fv.w;
    }
    __syncthreads();

    if (kb + 32 < S_DIM)
      __builtin_prefetch(omask + ((size_t)h * S_DIM + qb + w * 16 + m) * S_DIM + kb + 32, 0, 0);

    // scores for two 16-wide k tiles, mask+rowsum in f32, spill bf16 to lp
    for (int n = 0; n < 2; ++n) {
      const __bf16* krow = &lk[(16 * n + m) * 68];          // B = K^T, lane = k col
      Frag bK0, bK1;
      for (int j = 0; j < 8; ++j) {
        bK0.u32[j] = lds32(krow, (j & 3) + 8 * (j >> 2) + half * 4);             // d 0..31
        bK1.u32[j] = lds32(krow, 16 + (j & 3) + 8 * (j >> 2) + half * 4);        // d 32..63
      }
      v8f c;
      for (int i = 0; i < 8; ++i) c[i] = 0.f;
      c = __builtin_amdgcn_wmma_f32_16x16x32_bf16(false, aQ0.v, false, bK0.v, (short)0, c, false, false);
      c = __builtin_amdgcn_wmma_f32_16x16x32_bf16(false, aQ1.v, false, bK1.v, (short)0, c, false, false);

      const size_t obase =
          ((size_t)h * S_DIM + (qb + w * 16 + 8 * half)) * S_DIM + kb + 16 * n + m;
      for (int r = 0; r < 8; ++r) {
        float mv  = omask[obase + (size_t)r * S_DIM];
        float val = c[r] * mv;            // rs[q] already folded into Q
        srow[r] += val;
        lp[(w * 16 + r + 8 * half) * 34 + 16 * n + m] = (__bf16)val;
      }
    }

    // masked scores back out of LDS in A layout
    Frag pA;
    {
      const __bf16* prow = &lp[(w * 16 + m) * 34];
      for (int j = 0; j < 4; ++j) {
        pA.u32[j]     = lds32(prow, half * 4 + j);
        pA.u32[4 + j] = lds32(prow, 8 + half * 4 + j);
      }
    }
    // out += P[16x32] @ V[32x16] per 16-wide d tile
    for (int dt = 0; dt < 4; ++dt) {
      const __bf16* vrow = &lv[(dt * 16 + m) * 36];
      Frag bV;
      for (int j = 0; j < 8; ++j) bV.u32[j] = lds32(vrow, (j & 3) + 8 * (j >> 2) + half * 4);
      acc[dt] = __builtin_amdgcn_wmma_f32_16x16x32_bf16(false, pA.v, false, bV.v, (short)0,
                                                        acc[dt], false, false);
    }
  }

  // epilogue: per-row denom = clip(|rowsum|,1), scale and store
  for (int r = 0; r < 8; ++r) {
    float s = srow[r];
    s += __shfl_xor(s, 1);
    s += __shfl_xor(s, 2);
    s += __shfl_xor(s, 4);
    s += __shfl_xor(s, 8);       // halves reduce independently (rows r and r+8)
    float den = __builtin_fabsf(s);
    den = den < 1.f ? 1.f : den;
    float inv = 1.f / den;
    float* orow = out + bhBase + (size_t)(qb + w * 16 + r + 8 * half) * D_DIM;
    for (int dt = 0; dt < 4; ++dt)
      orow[dt * 16 + m] = acc[dt][r] * inv;
  }
}

extern "C" void kernel_launch(void* const* d_in, const int* in_sizes, int n_in,
                              void* d_out, int out_size, void* d_ws, size_t ws_size,
                              hipStream_t stream) {
  (void)in_sizes; (void)n_in; (void)out_size; (void)ws_size;
  const float* q     = (const float*)d_in[0];
  const float* k     = (const float*)d_in[1];
  const float* v     = (const float*)d_in[2];
  const float* omask = (const float*)d_in[3];
  float* rs = (float*)d_ws;  // H*S floats = 128 KB of workspace

  ParallelRetention_rowscale<<<dim3((H_DIM * S_DIM) / 8), 256, 0, stream>>>(omask, rs);
  ParallelRetention_main<<<dim3(S_DIM / 128, B_DIM * H_DIM), 256, 0, stream>>>(
      q, k, v, omask, rs, (float*)d_out);
}
